// GATModel_61349312856090
// MI455X (gfx1250) — matrix operand order repaired and checked by reading, not compiled, not run on hardware
//
#include <hip/hip_runtime.h>
#include <hip/hip_bf16.h>

// ---------------------------------------------------------------------------
// GAT forward for MI455X (gfx1250, wave32, WMMA).
//  - Dense phases (x@W, attention u-projections, MLP) use
//    v_wmma_f32_16x16x32_bf16, LDS-staged tiles, b128 LDS fragment loads.
//  - Edge phase uses factorized attention: e = v . leaky(u_src[s]+u_dst[d]),
//    u tables kept in bf16 (both fit in the 192MB L2), segment softmax via
//    uint-encoded atomicMax + f32 atomicAdd, aggregation via f32 atomic scatter.
// ---------------------------------------------------------------------------

typedef __attribute__((ext_vector_type(16))) __bf16 v16bf;
typedef __attribute__((ext_vector_type(8)))  __bf16 v8bf;
typedef __attribute__((ext_vector_type(8)))  float  v8f;

#define NEG_SLOPE 0.2f
#define HEADS 4
#define ATT   128

__device__ __forceinline__ __bf16 f32_to_bf16(float f) {
  __hip_bfloat16 h = __float2bfloat16(f);
  return *reinterpret_cast<__bf16*>(&h);
}

// order-preserving float -> uint encoding for atomicMax-based segment max
__device__ __forceinline__ unsigned enc_f32(float f) {
  unsigned u = __float_as_uint(f);
  return (u & 0x80000000u) ? ~u : (u | 0x80000000u);
}
__device__ __forceinline__ float dec_f32(unsigned e) {
  unsigned u = (e & 0x80000000u) ? (e & 0x7fffffffu) : ~e;
  return __uint_as_float(u);
}

// ---------------------------------------------------------------------------
__global__ void zero_b32(unsigned* __restrict__ p, long long n) {
  long long i = (long long)blockIdx.x * blockDim.x + threadIdx.x;
  if (i < n) p[i] = 0u;
}

// ---------------------------------------------------------------------------
// GEMM: C[M,Ncols] = act(A[M,K] @ B[K,Ncols] + bias),  K in {32,128} (templated,
// fully unrolled). A,B fp32 -> bf16 in LDS; f32 accumulation via WMMA.
// Block = 128 threads (4 wave32) -> 32x32 C tile, one 16x16 WMMA tile per wave.
// LDS: sA row-major [m][k]; sB transposed [n][k] so every lane's fragment is
// two contiguous 16B runs -> ds_load_b128 (no scalar ds_load_u16 storms).
// act: 0=none, 1=relu.  out_bf16: write C as bf16 instead of f32.
// ---------------------------------------------------------------------------
template <int K>
__global__ __launch_bounds__(128)
void gemm_wmma_bf16(const float* __restrict__ A, int lda,
                    const float* __restrict__ B, int ldb,
                    void* __restrict__ Cout, int ldc,
                    const float* __restrict__ bias,
                    int M, int Ncols, int act, int out_bf16)
{
  __shared__ __bf16 sA[32][40];    // [m][k], rows padded to 80B (16B-aligned)
  __shared__ __bf16 sBt[32][40];   // [n][k], transposed

  const int tid  = threadIdx.x;
  const int wave = tid >> 5;
  const int lane = tid & 31;
  const int wm   = (wave >> 1) * 16;
  const int wn   = (wave & 1) * 16;
  const int n0   = blockIdx.x * 32;
  const int m0   = blockIdx.y * 32;
  const bool hi  = lane >= 16;
  const int l15  = lane & 15;

  // staging assignment: thread owns row r, 8-column chunk c0
  const int r  = tid >> 2;
  const int c0 = (tid & 3) * 8;
  const int gr = m0 + r;
  const bool arow_ok = gr < M;
  const int bc0 = n0 + c0;
  const bool bfull = (bc0 + 7) < Ncols;

  v8f acc = {};

#pragma unroll
  for (int k0 = 0; k0 < K; k0 += 32) {
    // ---- stage A tile (row-major, f32 -> bf16, one b128 store) ----
    {
      float av[8] = {0.f, 0.f, 0.f, 0.f, 0.f, 0.f, 0.f, 0.f};
      if (arow_ok) {
        const float4* pa = (const float4*)(A + (long long)gr * lda + k0 + c0);
        float4 f0 = pa[0], f1 = pa[1];
        av[0] = f0.x; av[1] = f0.y; av[2] = f0.z; av[3] = f0.w;
        av[4] = f1.x; av[5] = f1.y; av[6] = f1.z; av[7] = f1.w;
      }
      v8bf pk;
#pragma unroll
      for (int j = 0; j < 8; ++j) pk[j] = f32_to_bf16(av[j]);
      *(v8bf*)&sA[r][c0] = pk;
    }
    // ---- stage B tile transposed (coalesced global, scattered b16 LDS) ----
    {
      float bv[8] = {0.f, 0.f, 0.f, 0.f, 0.f, 0.f, 0.f, 0.f};
      const float* pB = B + (long long)(k0 + r) * ldb + bc0;
      if (bfull) {
        const float4* pb = (const float4*)pB;
        float4 f0 = pb[0], f1 = pb[1];
        bv[0] = f0.x; bv[1] = f0.y; bv[2] = f0.z; bv[3] = f0.w;
        bv[4] = f1.x; bv[5] = f1.y; bv[6] = f1.z; bv[7] = f1.w;
      } else {
#pragma unroll
        for (int j = 0; j < 8; ++j)
          if (bc0 + j < Ncols) bv[j] = pB[j];
      }
#pragma unroll
      for (int j = 0; j < 8; ++j) sBt[c0 + j][r] = f32_to_bf16(bv[j]);
    }
    __syncthreads();

    // ---- fragments: contiguous 16B LDS runs -> ds_load_b128 ----
    // A 16x32 layout: lanes 0-15 hold K={0..7,16..23}, lanes 16-31 K={8..15,24..31}
    const int kb = hi ? 8 : 0;
    const v8bf aL = *(const v8bf*)&sA[wm + l15][kb];
    const v8bf aH = *(const v8bf*)&sA[wm + l15][kb + 16];
    const v16bf af = __builtin_shufflevector(aL, aH,
        0, 1, 2, 3, 4, 5, 6, 7, 8, 9, 10, 11, 12, 13, 14, 15);
    // B 32x16 layout: lane column N=l15; lanes 0-15 K={0..7,16..23}, lanes 16-31 K={8..15,24..31}
    const int kl = hi ? 8 : 0;
    const v8bf bL = *(const v8bf*)&sBt[wn + l15][kl];
    const v8bf bH = *(const v8bf*)&sBt[wn + l15][kl + 16];
    const v16bf bf_ = __builtin_shufflevector(bL, bH,
        0, 1, 2, 3, 4, 5, 6, 7, 8, 9, 10, 11, 12, 13, 14, 15);

    acc = __builtin_amdgcn_wmma_f32_16x16x32_bf16(false, af, false, bf_,
                                                  (short)0, acc, false, false);
    __syncthreads();
  }

  // C/D layout: VGPR i -> row (i + (hi?8:0)), col = lane&15
  const int col   = n0 + wn + l15;
  const int rbase = m0 + wm + (hi ? 8 : 0);
  const float bv  = (bias && col < Ncols) ? bias[col] : 0.f;
#pragma unroll
  for (int i = 0; i < 8; ++i) {
    int row = rbase + i;
    if (row < M && col < Ncols) {
      float v = acc[i] + bv;
      if (act == 1) v = fmaxf(v, 0.f);
      if (out_bf16)
        ((__hip_bfloat16*)Cout)[(long long)row * ldc + col] = __float2bfloat16(v);
      else
        ((float*)Cout)[(long long)row * ldc + col] = v;
    }
  }
}

// ---------------------------------------------------------------------------
// Edge attention scores: one wave per (edge, head).
// e = sum_a leaky(u_src[s,h,a] + u_dst[d,h,a]) * att_v[h,a]; atomicMax into m.
// u gathers are 8B uint2 loads; bf16 decoded exactly via <<16 bit tricks.
// ---------------------------------------------------------------------------
__global__ __launch_bounds__(256)
void edge_scores_kernel(const long long* __restrict__ src,
                        const long long* __restrict__ dst,
                        long long Ereal, long long Etot,
                        const __hip_bfloat16* __restrict__ us,
                        const __hip_bfloat16* __restrict__ ud,
                        const float* __restrict__ att_v,   // [4,128]
                        float* __restrict__ e_out,         // [Etot,4]
                        unsigned* __restrict__ m_enc)      // [N,4]
{
  long long eh = ((long long)blockIdx.x * blockDim.x + threadIdx.x) >> 5;
  if (eh >= Etot * HEADS) return;      // uniform across the wave
  long long eid = eh >> 2;
  int h = (int)(eh & 3);
  long long s, d;
  if (eid < Ereal) { s = src[eid]; d = dst[eid]; }
  else             { s = eid - Ereal; d = s; }

  int lane = threadIdx.x & 31;
  int a0 = lane * 4;
  if (lane == 0 && eid + 4096 < Ereal) {   // speculative index prefetch
    __builtin_prefetch(&src[eid + 4096], 0, 0);
    __builtin_prefetch(&dst[eid + 4096], 0, 0);
  }

  const uint2 va = *(const uint2*)(us + s * (HEADS * ATT) + h * ATT + a0);
  const uint2 vb = *(const uint2*)(ud + d * (HEADS * ATT) + h * ATT + a0);
  const float4 vv = *(const float4*)(att_v + h * ATT + a0);

  float zs[4];
  zs[0] = __uint_as_float(va.x << 16)         + __uint_as_float(vb.x << 16);
  zs[1] = __uint_as_float(va.x & 0xffff0000u) + __uint_as_float(vb.x & 0xffff0000u);
  zs[2] = __uint_as_float(va.y << 16)         + __uint_as_float(vb.y << 16);
  zs[3] = __uint_as_float(va.y & 0xffff0000u) + __uint_as_float(vb.y & 0xffff0000u);
  const float vvv[4] = {vv.x, vv.y, vv.z, vv.w};

  float sum = 0.f;
#pragma unroll
  for (int i = 0; i < 4; ++i) {
    float z = zs[i];
    z = (z > 0.f) ? z : NEG_SLOPE * z;
    sum += z * vvv[i];
  }
#pragma unroll
  for (int off = 16; off > 0; off >>= 1) sum += __shfl_xor(sum, off, 32);

  if (lane == 0) {
    e_out[eid * HEADS + h] = sum;
    atomicMax(&m_enc[d * HEADS + h], enc_f32(sum));
  }
}

// ---------------------------------------------------------------------------
__global__ __launch_bounds__(256)
void edge_exp_kernel(const long long* __restrict__ dst,
                     long long Ereal, long long Etot,
                     float* __restrict__ e_out,
                     const unsigned* __restrict__ m_enc,
                     float* __restrict__ den)
{
  long long idx = (long long)blockIdx.x * blockDim.x + threadIdx.x;
  if (idx >= Etot * HEADS) return;
  long long eid = idx >> 2;
  int h = (int)(idx & 3);
  long long d = (eid < Ereal) ? dst[eid] : (eid - Ereal);
  float m  = dec_f32(m_enc[d * HEADS + h]);
  float ex = __expf(e_out[idx] - m);
  e_out[idx] = ex;
  atomicAdd(&den[d * HEADS + h], ex);
}

// ---------------------------------------------------------------------------
// out[d, c0..c0+3] += alpha(e,h) * h[s, c0..c0+3]   (32 threads per edge)
// ---------------------------------------------------------------------------
__global__ __launch_bounds__(256)
void aggregate_kernel(const long long* __restrict__ src,
                      const long long* __restrict__ dst,
                      long long Ereal, long long Etot,
                      const float* __restrict__ e_out,
                      const float* __restrict__ den,
                      const float* __restrict__ hbuf,
                      float* __restrict__ xout)
{
  long long idx = (long long)blockIdx.x * blockDim.x + threadIdx.x;
  if (idx >= Etot * 32) return;
  long long eid = idx >> 5;
  int q  = (int)(idx & 31);
  int c0 = q * 4;
  int h  = q >> 3;
  long long s, d;
  if (eid < Ereal) { s = src[eid]; d = dst[eid]; }
  else             { s = eid - Ereal; d = s; }
  float ex = e_out[eid * HEADS + h];
  float dn = den[d * HEADS + h];
  float alpha = ex / (dn + 1e-16f);
  const float4 hv = *(const float4*)&hbuf[s * 128 + c0];
  float* o = &xout[d * 128 + c0];
  atomicAdd(o + 0, alpha * hv.x);
  atomicAdd(o + 1, alpha * hv.y);
  atomicAdd(o + 2, alpha * hv.z);
  atomicAdd(o + 3, alpha * hv.w);
}

// ---------------------------------------------------------------------------
__global__ __launch_bounds__(256)
void sigmoid_kernel(float* __restrict__ x, long long n) {
  long long i = (long long)blockIdx.x * blockDim.x + threadIdx.x;
  if (i < n) x[i] = 1.f / (1.f + __expf(-x[i]));
}

// ---------------------------------------------------------------------------
static void run_gemm(hipStream_t st, const float* A, int lda, const float* B,
                     int ldb, void* C, int ldc, const float* bias,
                     int M, int N, int K, int act, int out_bf16)
{
  dim3 grid((N + 31) / 32, (M + 31) / 32);
  if (K == 32)
    gemm_wmma_bf16<32><<<grid, 128, 0, st>>>(A, lda, B, ldb, C, ldc, bias,
                                             M, N, act, out_bf16);
  else
    gemm_wmma_bf16<128><<<grid, 128, 0, st>>>(A, lda, B, ldb, C, ldc, bias,
                                              M, N, act, out_bf16);
}

extern "C" void kernel_launch(void* const* d_in, const int* in_sizes, int n_in,
                              void* d_out, int out_size, void* d_ws, size_t ws_size,
                              hipStream_t stream)
{
  (void)n_in; (void)out_size; (void)ws_size;
  const float*     x0  = (const float*)d_in[0];
  const long long* ei  = (const long long*)d_in[1];
  const float*     W1  = (const float*)d_in[2];
  const float*     aw1 = (const float*)d_in[3];
  const float*     av1 = (const float*)d_in[4];
  const float*     W2  = (const float*)d_in[5];
  const float*     aw2 = (const float*)d_in[6];
  const float*     av2 = (const float*)d_in[7];
  const float*     mw1 = (const float*)d_in[8];
  const float*     mb1 = (const float*)d_in[9];
  const float*     mw2 = (const float*)d_in[10];
  const float*     mb2 = (const float*)d_in[11];

  const long long N    = in_sizes[0] / 128;
  const long long E    = in_sizes[1] / 2;
  const long long Etot = E + N;                // edges + self loops
  const int outN       = in_sizes[10] / 128;   // mlp_w2: [128, outN]

  // ---- workspace carve (256B aligned) ----
  char* p = (char*)d_ws;
  auto carve = [&](size_t bytes) -> char* {
    char* rr = p; p += (bytes + 255) & ~(size_t)255; return rr;
  };
  float*           h_buf = (float*)carve((size_t)N * 128 * 4);
  __hip_bfloat16*  u_s   = (__hip_bfloat16*)carve((size_t)N * 512 * 2);
  __hip_bfloat16*  u_d   = (__hip_bfloat16*)carve((size_t)N * 512 * 2);
  float*           e_buf = (float*)carve((size_t)Etot * 4 * 4);
  unsigned*        m_buf = (unsigned*)carve((size_t)N * 4 * 4);
  float*           den   = (float*)carve((size_t)N * 4 * 4);
  float*           x1    = (float*)carve((size_t)N * 128 * 4);
  float*           x2    = (float*)u_s;   // alias: written only after last u_s read
  float*           tb    = h_buf;         // alias: MLP hidden reuses h_buf

  const long long* srcp = ei;
  const long long* dstp = ei + E;

  auto layer = [&](const float* xin, const float* W, const float* aw,
                   const float* av, float* xout) {
    // h = xin @ W      [N,128] x [128,128]
    run_gemm(stream, xin, 128, W, 128, h_buf, 128, nullptr, (int)N, 128, 128, 0, 0);
    // u_src / u_dst per head: [N,32] x [32,128] -> bf16 [N, 4*128]
    for (int h = 0; h < HEADS; ++h) {
      run_gemm(stream, h_buf + h * 32, 128, aw + (size_t)h * 64 * 128, 128,
               u_s + h * 128, 512, nullptr, (int)N, 128, 32, 0, 1);
      run_gemm(stream, h_buf + h * 32, 128, aw + (size_t)h * 64 * 128 + 32 * 128, 128,
               u_d + h * 128, 512, nullptr, (int)N, 128, 32, 0, 1);
    }
    const long long n4 = N * 4;
    zero_b32<<<(unsigned)((n4 + 255) / 256), 256, 0, stream>>>(m_buf, n4);
    zero_b32<<<(unsigned)((n4 + 255) / 256), 256, 0, stream>>>((unsigned*)den, n4);

    const long long thr = Etot * 4 * 32;   // one wave per (edge, head)
    edge_scores_kernel<<<(unsigned)((thr + 255) / 256), 256, 0, stream>>>(
        srcp, dstp, E, Etot, u_s, u_d, av, e_buf, m_buf);
    const long long ehn = Etot * 4;
    edge_exp_kernel<<<(unsigned)((ehn + 255) / 256), 256, 0, stream>>>(
        dstp, E, Etot, e_buf, m_buf, den);

    const long long nout = N * 128;
    zero_b32<<<(unsigned)((nout + 255) / 256), 256, 0, stream>>>((unsigned*)xout, nout);
    const long long at = Etot * 32;
    aggregate_kernel<<<(unsigned)((at + 255) / 256), 256, 0, stream>>>(
        srcp, dstp, E, Etot, e_buf, den, h_buf, xout);
    sigmoid_kernel<<<(unsigned)((nout + 255) / 256), 256, 0, stream>>>(xout, nout);
  };

  layer(x0, W1, aw1, av1, x1);
  layer(x1, W2, aw2, av2, x2);

  // MLP head
  run_gemm(stream, x2, 128, mw1, 128, tb, 128, mb1, (int)N, 128, 128, 1, 0);
  run_gemm(stream, tb, 128, mw2, outN, d_out, outN, mb2, (int)N, outN, 128, 0, 0);
}
